// Model_2_1_Donor_34153579938662
// MI455X (gfx1250) — matrix operand order repaired and checked by reading, not compiled, not run on hardware
//
#include <hip/hip_runtime.h>

#define BATCH 512
#define TT    2048
#define HID   16

typedef __attribute__((ext_vector_type(16))) _Float16 v16h;
typedef __attribute__((ext_vector_type(8)))  _Float16 v8h;
typedef __attribute__((ext_vector_type(8)))  float    v8f;

__device__ __forceinline__ float fast_sig(float x) {
  // 1 / (1 + e^-x) via v_exp_f32 (exp2) + v_rcp_f32
  return __builtin_amdgcn_rcpf(1.0f + __builtin_amdgcn_exp2f(-1.44269504f * x));
}
__device__ __forceinline__ float fast_tanh(float x) {
  // tanh(x) = 1 - 2/(e^{2x}+1)
  return 1.0f - 2.0f * __builtin_amdgcn_rcpf(1.0f + __builtin_amdgcn_exp2f(2.88539008f * x));
}
__device__ __forceinline__ float lrelu(float x) { return x > 0.0f ? x : 0.01f * x; }

// ---------------------------------------------------------------------------
// One wave owns a 16-row batch tile and runs the full T-step recurrence.
//   gates[16,64] = A_x[16,32] * B_x[32,64]  +  A_h[16,32] * B_h[32,64]  + bias
// done as 4 N-tiles of v_wmma_f32_16x16x32_f16 (x) + 4 more (h).
// h/c state stays in registers in the C/D accumulator layout
// (lane = gate column, VGPR = batch row); the per-step h D->A relayout goes
// through 1KB of wave-private LDS.
// ---------------------------------------------------------------------------
__global__ __launch_bounds__(32) void lstm_scan(
    const float*    __restrict__ x0,   // layer-0 input [B,T,2] fp32, or nullptr
    const _Float16* __restrict__ xh,   // deeper input  [B,T,32] fp16, or nullptr
    const float* __restrict__ WihF, const float* __restrict__ WhhF,
    const float* __restrict__ bihF, const float* __restrict__ bhhF,
    const float* __restrict__ WihB, const float* __restrict__ WhhB,
    const float* __restrict__ bihB, const float* __restrict__ bhhB,
    _Float16* __restrict__ xout,       // [B,T,32] fp16 (fwd half / bwd half), or nullptr
    float*    __restrict__ hcOut,      // [B,32] fp32 {h | c}, or nullptr (final mode)
    int I, int finalMode)
{
  __shared__ float hbuf[16 * 17];      // 17-float row pitch to dodge bank conflicts
  const int lane  = threadIdx.x;       // wave32
  const int nCol  = lane & 15;         // D-layout column / A-layout row
  const int kHalf = lane >> 4;         // which K half this lane holds
  const int b0    = blockIdx.x * 16;   // batch tile base
  const int dir   = blockIdx.y;        // 0 = fwd, 1 = bwd
  const bool rev  = (dir == 1);

  const float* Wih = dir ? WihB : WihF;
  const float* Whh = dir ? WhhB : WhhF;
  const float* bih = dir ? bihB : bihF;
  const float* bhh = dir ? bhhB : bhhF;

  // ---- Build resident f16 B operands (K x N tiles, K padded to 32) --------
  // B layout (16-bit, 32x16): element j of v16h = K = kHalf*8 + j (j<8), 16 + kHalf*8 + (j-8) (j>=8), N = nCol.
  v16h Bx[4], Bh[4];
  float bias[4];
  #pragma unroll
  for (int nt = 0; nt < 4; ++nt) {
    const int n = nt * 16 + nCol;      // global gate column (i,f,g,o blocks)
    #pragma unroll
    for (int j = 0; j < 16; ++j) {
      const int k = (j < 8) ? (kHalf * 8 + j) : (16 + kHalf * 8 + (j - 8));
      Bx[nt][j] = (_Float16)((k < I)   ? Wih[n * I   + k] : 0.0f);
      Bh[nt][j] = (_Float16)((k < HID) ? Whh[n * HID + k] : 0.0f);
    }
    bias[nt] = bih[n] + bhh[n];
  }

  // A layout (16-bit, 16x32): row M = lane&15; lanes 0-15 hold K 0..7 (els 0-7)
  // and K 16..23 (els 8-15); lanes 16-31 hold K 8..15 and 24..31.
  v16h Ah = {};                        // h state, K 16..31 stay zero (H=16)
  float cst[8], hst[8];
  #pragma unroll
  for (int r = 0; r < 8; ++r) { cst[r] = 0.0f; hst[r] = 0.0f; }

  const int m = nCol;                  // A-matrix row owned by this lane

  for (int s = 0; s < TT; ++s) {
    const int t = rev ? (TT - 1 - s) : s;

    // ---- A operand for x_t ------------------------------------------------
    v16h Ax = {};
    if (x0) {                          // layer 0: I = 2, fp32 source
      if (kHalf == 0) {
        const float* xp = x0 + ((size_t)(b0 + m) * TT + t) * 2;
        Ax[0] = (_Float16)xp[0];
        Ax[1] = (_Float16)xp[1];
      }
    } else {                           // deeper layers: I = 32, fp16 source
      const _Float16* xp = xh + ((size_t)(b0 + m) * TT + t) * 32 + kHalf * 8;
      v8h lo = *(const v8h*)xp;        // K slice [kHalf*8 .. +7]
      v8h hi = *(const v8h*)(xp + 16); // K slice [16+kHalf*8 .. +7]
      #pragma unroll
      for (int j = 0; j < 8; ++j) { Ax[j] = lo[j]; Ax[8 + j] = hi[j]; }
      const int tp = rev ? (t >= 8 ? t - 8 : 0) : (t + 8 < TT ? t + 8 : TT - 1);
      __builtin_prefetch(xh + ((size_t)(b0 + m) * TT + tp) * 32, 0, 0);
    }

    // ---- gates = x_t*WihT + h*WhhT + bias : 8 WMMAs -----------------------
    v8f g[4];
    #pragma unroll
    for (int nt = 0; nt < 4; ++nt) {
      v8f a;
      #pragma unroll
      for (int r = 0; r < 8; ++r) a[r] = bias[nt];
      a = __builtin_amdgcn_wmma_f32_16x16x32_f16(false, Ax, false, Bx[nt],
                                                 (short)0, a, false, false);
      a = __builtin_amdgcn_wmma_f32_16x16x32_f16(false, Ah, false, Bh[nt],
                                                 (short)0, a, false, false);
      g[nt] = a;
    }

    // ---- pointwise LSTM cell (tile 0..3 = i,f,g,o; same (M,N) per lane) ---
    #pragma unroll
    for (int r = 0; r < 8; ++r) {
      const float ii = fast_sig(g[0][r]);
      const float ff = fast_sig(g[1][r]);
      const float gg = fast_tanh(g[2][r]);
      const float oo = fast_sig(g[3][r]);
      const float cr = ff * cst[r] + ii * gg;
      cst[r] = cr;
      hst[r] = oo * fast_tanh(cr);
    }

    // ---- h: D layout -> LDS -> A layout; also emit the layer output ------
    #pragma unroll
    for (int r = 0; r < 8; ++r) {
      const int M = r + 8 * kHalf;     // batch row
      hbuf[M * 17 + nCol] = hst[r];
      if (xout)
        xout[((size_t)(b0 + M) * TT + t) * 32 + dir * 16 + nCol] = (_Float16)hst[r];
    }
    __syncthreads();                   // 1-wave WG: S_NOP + compiler fence
    v16h nAh = {};
    #pragma unroll
    for (int j = 0; j < 8; ++j)
      nAh[j] = (_Float16)hbuf[m * 17 + kHalf * 8 + j];
    Ah = nAh;
    __syncthreads();
  }

  if (finalMode && hcOut) {
    #pragma unroll
    for (int r = 0; r < 8; ++r) {
      const int M = r + 8 * kHalf;
      hcOut[(size_t)(b0 + M) * 32 + nCol]      = hst[r];
      hcOut[(size_t)(b0 + M) * 32 + 16 + nCol] = cst[r];
    }
  }
}

// ---------------------------------------------------------------------------
// MLP head: out = [c3(...), r3(...)] on concat(hT, hT, cT). One thread per
// batch row; weight reads are wave-uniform -> scalar loads.
// ---------------------------------------------------------------------------
__global__ __launch_bounds__(256) void head_kernel(
    const float* __restrict__ hc,
    const float* __restrict__ W1,  const float* __restrict__ b1,
    const float* __restrict__ W2,  const float* __restrict__ b2,
    const float* __restrict__ W3,  const float* __restrict__ b3,
    const float* __restrict__ Wc1, const float* __restrict__ bc1,
    const float* __restrict__ Wc2, const float* __restrict__ bc2,
    const float* __restrict__ Wc3, const float* __restrict__ bc3,
    const float* __restrict__ Wr1, const float* __restrict__ br1,
    const float* __restrict__ Wr2, const float* __restrict__ br2,
    const float* __restrict__ Wr3, const float* __restrict__ br3,
    float* __restrict__ out)
{
  const int b = blockIdx.x * blockDim.x + threadIdx.x;
  if (b >= BATCH) return;

  float in48[48];
  #pragma unroll
  for (int j = 0; j < 16; ++j) {
    const float h = hc[(size_t)b * 32 + j];
    in48[j] = h;                        // o2[:, -1, :] == hT
    in48[16 + j] = h;                   // hT
    in48[32 + j] = hc[(size_t)b * 32 + 16 + j];  // cT
  }
  float t1[32], t2[32];
  for (int j = 0; j < 32; ++j) {
    float s = b1[j];
    for (int k = 0; k < 48; ++k) s += W1[j * 48 + k] * in48[k];
    t1[j] = lrelu(s);
  }
  for (int j = 0; j < 32; ++j) {
    float s = b2[j];
    for (int k = 0; k < 32; ++k) s += W2[j * 32 + k] * t1[k];
    t2[j] = lrelu(s);
  }
  for (int j = 0; j < 32; ++j) {
    float s = b3[j];
    for (int k = 0; k < 32; ++k) s += W3[j * 32 + k] * t2[k];
    t1[j] = lrelu(s);                   // t1 = ld3 output ("out")
  }

  float u[64], v[64];
  // chi branch
  for (int j = 0; j < 64; ++j) {
    float s = bc1[j];
    for (int k = 0; k < 32; ++k) s += Wc1[j * 32 + k] * t1[k];
    u[j] = lrelu(s);
  }
  for (int j = 0; j < 64; ++j) {
    float s = bc2[j];
    for (int k = 0; k < 64; ++k) s += Wc2[j * 64 + k] * u[k];
    v[j] = lrelu(s);
  }
  float chi = bc3[0];
  for (int k = 0; k < 64; ++k) chi += Wc3[k] * v[k];
  // rp branch
  for (int j = 0; j < 64; ++j) {
    float s = br1[j];
    for (int k = 0; k < 32; ++k) s += Wr1[j * 32 + k] * t1[k];
    u[j] = lrelu(s);
  }
  for (int j = 0; j < 64; ++j) {
    float s = br2[j];
    for (int k = 0; k < 64; ++k) s += Wr2[j * 64 + k] * u[k];
    v[j] = lrelu(s);
  }
  float rp = br3[0];
  for (int k = 0; k < 64; ++k) rp += Wr3[k] * v[k];

  out[(size_t)b * 2 + 0] = chi;
  out[(size_t)b * 2 + 1] = rp;
}

// ---------------------------------------------------------------------------
// Input order (setup_inputs dict order, recursively flattened):
// 0 Traces, 1 AuxData,
// lstm1[l].{fwd,bwd}.{Wih,Whh,bih,bhh}: fwd base 2+8l, bwd base 6+8l (l=0..2)
// lstm2: 26..29
// ld1 30,31  ld2 32,33  ld3 34,35
// c1 36,37  c2 38,39  c3 40,41   r1 42,43  r2 44,45  r3 46,47
// ---------------------------------------------------------------------------
extern "C" void kernel_launch(void* const* d_in, const int* in_sizes, int n_in,
                              void* d_out, int out_size, void* d_ws, size_t ws_size,
                              hipStream_t stream)
{
  (void)in_sizes; (void)n_in; (void)out_size; (void)ws_size;
  auto F = [&](int i) { return (const float*)d_in[i]; };
  const float* Traces = F(0);

  const size_t xe = (size_t)BATCH * TT * 32;        // elems per activation buffer
  _Float16* xA = (_Float16*)d_ws;                   // 64 MB
  _Float16* xB = xA + xe;                           // 64 MB
  float*    hc = (float*)(xB + xe);                 // [B,32] hT|cT

  dim3 blk(32);
  dim3 gBi(BATCH / 16, 2);                          // fwd + bwd concurrently
  dim3 gUni(BATCH / 16, 1);

  // lstm1 layer 0: fp32 [B,T,2] -> xA
  lstm_scan<<<gBi, blk, 0, stream>>>(Traces, nullptr,
      F(2), F(3), F(4), F(5),  F(6), F(7), F(8), F(9),
      xA, nullptr, 2, 0);
  // lstm1 layer 1: xA -> xB
  lstm_scan<<<gBi, blk, 0, stream>>>(nullptr, xA,
      F(10), F(11), F(12), F(13),  F(14), F(15), F(16), F(17),
      xB, nullptr, 32, 0);
  // lstm1 layer 2: xB -> xA
  lstm_scan<<<gBi, blk, 0, stream>>>(nullptr, xB,
      F(18), F(19), F(20), F(21),  F(22), F(23), F(24), F(25),
      xA, nullptr, 32, 0);
  // lstm2 (fwd only): xA -> hT/cT
  lstm_scan<<<gUni, blk, 0, stream>>>(nullptr, xA,
      F(26), F(27), F(28), F(29),  F(26), F(27), F(28), F(29),
      nullptr, hc, 32, 1);
  // head
  head_kernel<<<BATCH / 256, 256, 0, stream>>>(hc,
      F(30), F(31), F(32), F(33), F(34), F(35),
      F(36), F(37), F(38), F(39), F(40), F(41),
      F(42), F(43), F(44), F(45), F(46), F(47),
      (float*)d_out);
}
